// MultiHeadAttention_34935263986240
// MI455X (gfx1250) — compile-verified, matrix operand order checked by reading, not run on hardware
//
#include <hip/hip_runtime.h>

typedef __attribute__((ext_vector_type(16))) _Float16 v16h;
typedef __attribute__((ext_vector_type(8)))  _Float16 h8;
typedef __attribute__((ext_vector_type(8)))  float    v8f;
typedef __attribute__((ext_vector_type(4)))  float    f32x4;

namespace {
constexpr int kD     = 1024;
constexpr int kHeads = 16;
constexpr int kDh    = 64;
constexpr int kSeq   = 2048;
constexpr int kBatch = 2;
constexpr int kTok   = kBatch * kSeq;      // 4096
constexpr int kPitch = 40;                 // LDS pitch in halves (32 + 8 pad)
constexpr int kBufHalves = 128 * kPitch;   // 5120 halves = 10240 B per tile buffer
}

// ---------------------------------------------------------------------------
// CDNA5 async global->LDS copy (ASYNCcnt-tracked), 16 bytes per lane.
// lds_off: byte offset within the workgroup's LDS allocation (flat-shared
// pointer truncated to 32 bits).  gptr: per-lane global address.
// ---------------------------------------------------------------------------
__device__ __forceinline__ void async_b128(unsigned int lds_off, const _Float16* gptr) {
  asm volatile("global_load_async_to_lds_b128 %0, %1, off"
               :: "v"(lds_off), "v"(gptr)
               : "memory");
}
__device__ __forceinline__ void wait_async_le4() {
  asm volatile("s_wait_asynccnt 4" ::: "memory");
}
__device__ __forceinline__ void wait_async_0() {
  asm volatile("s_wait_asynccnt 0" ::: "memory");
}

// Build a 16-half A/B fragment for wmma_f32_16x16x32_f16.
// Elements 0..7 hold k = 8*(lane/16)+0..7, elements 8..15 hold k+16.
// Caller passes row pointer already offset by 8*(lane/16).
__device__ __forceinline__ v16h make_frag(const _Float16* p) {
  h8 lo = *(const h8*)p;
  h8 hi = *(const h8*)(p + 16);
  return __builtin_shufflevector(lo, hi, 0,1,2,3,4,5,6,7,8,9,10,11,12,13,14,15);
}

__device__ __forceinline__ v8f wmma16(v16h a, v16h b, v8f c) {
  return __builtin_amdgcn_wmma_f32_16x16x32_f16(false, a, false, b, (short)0, c, false, false);
}

// ---------------------------------------------------------------------------
// Kernel 0: elementwise f32 -> f16 convert (one-shot; removes all per-tile
// conversion work from the GEMMs).  8 elems/thread, n multiple of 2048.
// ---------------------------------------------------------------------------
__global__ __launch_bounds__(256) void cvt_f16_kernel(const float* __restrict__ src,
                                                      _Float16* __restrict__ dst) {
  const size_t i = ((size_t)blockIdx.x * 256 + threadIdx.x) * 8;
  f32x4 a = *(const f32x4*)(src + i);
  f32x4 b = *(const f32x4*)(src + i + 4);
  h8 r;
  r[0] = (_Float16)a[0]; r[1] = (_Float16)a[1]; r[2] = (_Float16)a[2]; r[3] = (_Float16)a[3];
  r[4] = (_Float16)b[0]; r[5] = (_Float16)b[1]; r[6] = (_Float16)b[2]; r[7] = (_Float16)b[3];
  *(h8*)(dst + i) = r;
}

// ---------------------------------------------------------------------------
// Kernel A: fused Q/K/V projection, f16 in, async double-buffered LDS.
// grid=(N/128, M/128, 3), block=256 (8 waves, each 64x32 of a 128x128 tile).
// Q,K stored (B,H,L,Dh); V stored transposed (B,H,Dh,L).
// ---------------------------------------------------------------------------
__global__ __launch_bounds__(256) void qkv_proj_kernel(
    const _Float16* __restrict__ xq16, const _Float16* __restrict__ xkv16,
    const _Float16* __restrict__ wq16, const _Float16* __restrict__ wk16,
    const _Float16* __restrict__ wv16,
    const float* __restrict__ bq, const float* __restrict__ bk,
    const float* __restrict__ bv,
    _Float16* __restrict__ Qo, _Float16* __restrict__ Ko, _Float16* __restrict__ Vto)
{
  const int which = blockIdx.z;                       // 0=Q, 1=K, 2=V
  const _Float16* x    = (which == 0) ? xq16 : xkv16;
  const _Float16* w    = (which == 0) ? wq16 : (which == 1 ? wk16 : wv16);
  const float*    bias = (which == 0) ? bq   : (which == 1 ? bk   : bv);

  __shared__ __align__(16) _Float16 As[2][kBufHalves];
  __shared__ __align__(16) _Float16 Bs[2][kBufHalves];

  const int t    = threadIdx.x;
  const int lane = t & 31;
  const int wave = t >> 5;
  const int lrow = lane & 15;
  const int lhi  = lane >> 4;
  const int mwg  = blockIdx.y * 128;
  const int nwg  = blockIdx.x * 128;
  const int moff = (wave >> 2) * 64;
  const int noff = (wave & 3) * 32;
  const int srow  = t >> 1;               // 0..127
  const int chunk = (t & 1) * 16;         // 0 / 16 (halves)

  // Per-thread staging addresses.
  const _Float16* gA = x + (size_t)(mwg + srow) * kD + chunk;
  const _Float16* gB = w + (size_t)(nwg + srow) * kD + chunk;
  const unsigned int ldsA0 = (unsigned int)(uintptr_t)&As[0][srow * kPitch + chunk];
  const unsigned int ldsB0 = (unsigned int)(uintptr_t)&Bs[0][srow * kPitch + chunk];
  const unsigned int bufB  = kBufHalves * 2;   // bytes per buffer

  v8f acc[4][2];
#pragma unroll
  for (int i = 0; i < 4; ++i)
#pragma unroll
    for (int j = 0; j < 2; ++j)
#pragma unroll
      for (int r = 0; r < 8; ++r) acc[i][j][r] = 0.0f;

  // Preload tile 0 into buffer 0.
  async_b128(ldsA0,      gA);
  async_b128(ldsA0 + 16, gA + 8);
  async_b128(ldsB0,      gB);
  async_b128(ldsB0 + 16, gB + 8);

  for (int k0 = 0; k0 < kD; k0 += 32) {
    const int buf = (k0 >> 5) & 1;
    if (k0 + 32 < kD) {  // issue next tile into other buffer
      const unsigned int nb = (buf ^ 1) * bufB;
      async_b128(ldsA0 + nb,      gA + k0 + 32);
      async_b128(ldsA0 + nb + 16, gA + k0 + 40);
      async_b128(ldsB0 + nb,      gB + k0 + 32);
      async_b128(ldsB0 + nb + 16, gB + k0 + 40);
      wait_async_le4();             // current tile's 4 copies have landed
    } else {
      wait_async_0();
    }
    __syncthreads();                // all waves' copies for `buf` visible

    v16h afr[4], bfr[2];
#pragma unroll
    for (int i = 0; i < 4; ++i)
      afr[i] = make_frag(&As[buf][(moff + 16 * i + lrow) * kPitch + 8 * lhi]);
#pragma unroll
    for (int j = 0; j < 2; ++j)
      bfr[j] = make_frag(&Bs[buf][(noff + 16 * j + lrow) * kPitch + 8 * lhi]);
#pragma unroll
    for (int i = 0; i < 4; ++i)
#pragma unroll
      for (int j = 0; j < 2; ++j)
        acc[i][j] = wmma16(afr[i], bfr[j], acc[i][j]);
    __syncthreads();                // reads done before buffer is re-staged
  }

  if (which == 2) {
#pragma unroll
    for (int j = 0; j < 2; ++j) {
      const int n = nwg + noff + 16 * j + lrow;
      const float bb = bias[n];
      const int h = n >> 6, dh = n & 63;
#pragma unroll
      for (int i = 0; i < 4; ++i)
#pragma unroll
        for (int r = 0; r < 8; ++r) {
          const int m = mwg + moff + 16 * i + r + 8 * lhi;
          const int b = m >> 11, l = m & (kSeq - 1);
          Vto[(((size_t)(b * kHeads + h)) * kDh + dh) * kSeq + l] =
              (_Float16)(acc[i][j][r] + bb);
        }
    }
  } else {
    _Float16* out = which ? Ko : Qo;
#pragma unroll
    for (int j = 0; j < 2; ++j) {
      const int n = nwg + noff + 16 * j + lrow;
      const float bb = bias[n];
      const int h = n >> 6, dh = n & 63;
#pragma unroll
      for (int i = 0; i < 4; ++i)
#pragma unroll
        for (int r = 0; r < 8; ++r) {
          const int m = mwg + moff + 16 * i + r + 8 * lhi;
          const int b = m >> 11, l = m & (kSeq - 1);
          out[(((size_t)(b * kHeads + h)) * kSeq + l) * kDh + dh] =
              (_Float16)(acc[i][j][r] + bb);
        }
    }
  }
}

// ---------------------------------------------------------------------------
// Kernel B: flash attention. grid=(Seq/128, B*H), block=256 (8 waves).
// Each wave owns 16 queries; computes S^T = K*Q^T so each lane owns one
// query's softmax state; O^T = V^T * P^T reuses score VGPRs as B-fragments.
// ---------------------------------------------------------------------------
__global__ __launch_bounds__(256) void attn_kernel(
    const _Float16* __restrict__ Q, const _Float16* __restrict__ K,
    const _Float16* __restrict__ Vt, _Float16* __restrict__ ctx)
{
  const int t    = threadIdx.x;
  const int lane = t & 31;
  const int wave = t >> 5;
  const int lrow = lane & 15;
  const int lhi  = lane >> 4;
  const int bh   = blockIdx.y;
  const int q0   = blockIdx.x * 128 + wave * 16;

  const _Float16* Qh = Q  + (size_t)bh * kSeq * kDh;
  const _Float16* Kh = K  + (size_t)bh * kSeq * kDh;
  const _Float16* Vh = Vt + (size_t)bh * kDh * kSeq;

  v16h qf[2];
#pragma unroll
  for (int s = 0; s < 2; ++s)
    qf[s] = make_frag(Qh + (size_t)(q0 + lrow) * kDh + s * 32 + 8 * lhi);

  v8f o[4];
#pragma unroll
  for (int i = 0; i < 4; ++i)
#pragma unroll
    for (int r = 0; r < 8; ++r) o[i][r] = 0.0f;

  float mrun = -1e30f, srun = 0.0f;
  const float sc = 0.125f;  // 1/sqrt(64)

  for (int kb = 0; kb < kSeq; kb += 32) {
    v8f st0, st1;
#pragma unroll
    for (int r = 0; r < 8; ++r) { st0[r] = 0.0f; st1[r] = 0.0f; }
#pragma unroll
    for (int s = 0; s < 2; ++s) {
      v16h kf0 = make_frag(Kh + (size_t)(kb + lrow) * kDh + s * 32 + 8 * lhi);
      v16h kf1 = make_frag(Kh + (size_t)(kb + 16 + lrow) * kDh + s * 32 + 8 * lhi);
      st0 = wmma16(kf0, qf[s], st0);
      st1 = wmma16(kf1, qf[s], st1);
    }

    float p0[8], p1[8];
    float bm = -1e30f;
#pragma unroll
    for (int r = 0; r < 8; ++r) {
      p0[r] = st0[r] * sc;  bm = fmaxf(bm, p0[r]);
      p1[r] = st1[r] * sc;  bm = fmaxf(bm, p1[r]);
    }
    bm = fmaxf(bm, __shfl_xor(bm, 16, 32));
    const float mnew = fmaxf(mrun, bm);
    const float corr = __expf(mrun - mnew);
    float rs = 0.0f;
#pragma unroll
    for (int r = 0; r < 8; ++r) {
      p0[r] = __expf(p0[r] - mnew);
      p1[r] = __expf(p1[r] - mnew);
      rs += p0[r] + p1[r];
    }
    rs += __shfl_xor(rs, 16, 32);
    srun = srun * corr + rs;
    mrun = mnew;
#pragma unroll
    for (int i = 0; i < 4; ++i)
#pragma unroll
      for (int r = 0; r < 8; ++r) o[i][r] *= corr;

    v16h pt;
#pragma unroll
    for (int r = 0; r < 8; ++r) {
      pt[r]     = (_Float16)p0[r];
      pt[r + 8] = (_Float16)p1[r];
    }

#pragma unroll
    for (int i = 0; i < 4; ++i) {
      v16h vf = make_frag(Vh + (size_t)(16 * i + lrow) * kSeq + kb + 8 * lhi);
      o[i] = wmma16(vf, pt, o[i]);
    }
  }

  const float inv = 1.0f / srun;
  const int b = bh >> 4, h = bh & 15;
  _Float16* dst = ctx + ((size_t)(b * kSeq + q0 + lrow)) * kD + h * kDh;
#pragma unroll
  for (int i = 0; i < 4; ++i) {
    h8 ov;
#pragma unroll
    for (int r = 0; r < 8; ++r) ov[r] = (_Float16)(o[i][r] * inv);
    *(h8*)(dst + 16 * i + 8 * lhi) = ov;
  }
}

// ---------------------------------------------------------------------------
// Kernel C: output projection. out = ctx @ wo^T + bo, f32 result.
// grid=(8, 32), block=256. Same async double-buffered structure as kernel A.
// ---------------------------------------------------------------------------
__global__ __launch_bounds__(256) void out_proj_kernel(
    const _Float16* __restrict__ ctx, const _Float16* __restrict__ wo16,
    const float* __restrict__ bo, float* __restrict__ out)
{
  __shared__ __align__(16) _Float16 As[2][kBufHalves];
  __shared__ __align__(16) _Float16 Bs[2][kBufHalves];

  const int t    = threadIdx.x;
  const int lane = t & 31;
  const int wave = t >> 5;
  const int lrow = lane & 15;
  const int lhi  = lane >> 4;
  const int mwg  = blockIdx.y * 128;
  const int nwg  = blockIdx.x * 128;
  const int moff = (wave >> 2) * 64;
  const int noff = (wave & 3) * 32;
  const int srow  = t >> 1;
  const int chunk = (t & 1) * 16;

  const _Float16* gA = ctx  + (size_t)(mwg + srow) * kD + chunk;
  const _Float16* gB = wo16 + (size_t)(nwg + srow) * kD + chunk;
  const unsigned int ldsA0 = (unsigned int)(uintptr_t)&As[0][srow * kPitch + chunk];
  const unsigned int ldsB0 = (unsigned int)(uintptr_t)&Bs[0][srow * kPitch + chunk];
  const unsigned int bufB  = kBufHalves * 2;

  v8f acc[4][2];
#pragma unroll
  for (int i = 0; i < 4; ++i)
#pragma unroll
    for (int j = 0; j < 2; ++j)
#pragma unroll
      for (int r = 0; r < 8; ++r) acc[i][j][r] = 0.0f;

  async_b128(ldsA0,      gA);
  async_b128(ldsA0 + 16, gA + 8);
  async_b128(ldsB0,      gB);
  async_b128(ldsB0 + 16, gB + 8);

  for (int k0 = 0; k0 < kD; k0 += 32) {
    const int buf = (k0 >> 5) & 1;
    if (k0 + 32 < kD) {
      const unsigned int nb = (buf ^ 1) * bufB;
      async_b128(ldsA0 + nb,      gA + k0 + 32);
      async_b128(ldsA0 + nb + 16, gA + k0 + 40);
      async_b128(ldsB0 + nb,      gB + k0 + 32);
      async_b128(ldsB0 + nb + 16, gB + k0 + 40);
      wait_async_le4();
    } else {
      wait_async_0();
    }
    __syncthreads();

    v16h afr[4], bfr[2];
#pragma unroll
    for (int i = 0; i < 4; ++i)
      afr[i] = make_frag(&As[buf][(moff + 16 * i + lrow) * kPitch + 8 * lhi]);
#pragma unroll
    for (int j = 0; j < 2; ++j)
      bfr[j] = make_frag(&Bs[buf][(noff + 16 * j + lrow) * kPitch + 8 * lhi]);
#pragma unroll
    for (int i = 0; i < 4; ++i)
#pragma unroll
      for (int j = 0; j < 2; ++j)
        acc[i][j] = wmma16(afr[i], bfr[j], acc[i][j]);
    __syncthreads();
  }

#pragma unroll
  for (int j = 0; j < 2; ++j) {
    const int n = nwg + noff + 16 * j + lrow;
    const float bb = bo[n];
#pragma unroll
    for (int i = 0; i < 4; ++i)
#pragma unroll
      for (int r = 0; r < 8; ++r) {
        const int m = mwg + moff + 16 * i + r + 8 * lhi;
        out[(size_t)m * kD + n] = acc[i][j][r] + bb;
      }
  }
}

extern "C" void kernel_launch(void* const* d_in, const int* in_sizes, int n_in,
                              void* d_out, int out_size, void* d_ws, size_t ws_size,
                              hipStream_t stream) {
  (void)in_sizes; (void)n_in; (void)out_size; (void)ws_size;
  const float* xq  = (const float*)d_in[0];
  const float* xkv = (const float*)d_in[1];
  const float* wq  = (const float*)d_in[2];
  const float* bq  = (const float*)d_in[3];
  const float* wk  = (const float*)d_in[4];
  const float* bk  = (const float*)d_in[5];
  const float* wv  = (const float*)d_in[6];
  const float* bv  = (const float*)d_in[7];
  const float* wo  = (const float*)d_in[8];
  const float* bo  = (const float*)d_in[9];

  const size_t xel = (size_t)kTok * kD;   // 4 Mi elems
  const size_t wel = (size_t)kD * kD;     // 1 Mi elems

  _Float16* xq16  = (_Float16*)d_ws;
  _Float16* xkv16 = xq16  + xel;
  _Float16* wq16  = xkv16 + xel;
  _Float16* wk16  = wq16  + wel;
  _Float16* wv16  = wk16  + wel;
  _Float16* wo16  = wv16  + wel;
  _Float16* Q     = wo16  + wel;
  _Float16* K     = Q     + xel;
  _Float16* Vt    = K     + xel;
  _Float16* ctx   = Vt    + xel;

  // One-shot f32 -> f16 conversion of activations and weights.
  cvt_f16_kernel<<<dim3((unsigned)(xel / 2048)), 256, 0, stream>>>(xq,  xq16);
  cvt_f16_kernel<<<dim3((unsigned)(xel / 2048)), 256, 0, stream>>>(xkv, xkv16);
  cvt_f16_kernel<<<dim3((unsigned)(wel / 2048)), 256, 0, stream>>>(wq,  wq16);
  cvt_f16_kernel<<<dim3((unsigned)(wel / 2048)), 256, 0, stream>>>(wk,  wk16);
  cvt_f16_kernel<<<dim3((unsigned)(wel / 2048)), 256, 0, stream>>>(wv,  wv16);
  cvt_f16_kernel<<<dim3((unsigned)(wel / 2048)), 256, 0, stream>>>(wo,  wo16);

  qkv_proj_kernel<<<dim3(kD / 128, kTok / 128, 3), 256, 0, stream>>>(
      xq16, xkv16, wq16, wk16, wv16, bq, bk, bv, Q, K, Vt);
  attn_kernel<<<dim3(kSeq / 128, kBatch * kHeads), 256, 0, stream>>>(Q, K, Vt, ctx);
  out_proj_kernel<<<dim3(kD / 128, kTok / 128), 256, 0, stream>>>(ctx, wo16, bo, (float*)d_out);
}
